// GCN5_51780125721120
// MI455X (gfx1250) — matrix operand order repaired and checked by reading, not compiled, not run on hardware
//
#include <hip/hip_runtime.h>
#include <hip/hip_bf16.h>
#include <math.h>

#define NNODES  20000
#define NEDGES  320000
#define NGRAPHS 50

typedef __attribute__((ext_vector_type(16))) __bf16 v16bf;
typedef __attribute__((ext_vector_type(8)))  float  v8f;

// pack two f32 -> one dword of bf16 pair (lowers to v_cvt_pk_bf16_f32)
__device__ __forceinline__ unsigned pkbf(float a, float b) {
  __hip_bfloat162 h = __float22bfloat162_rn(make_float2(a, b));
  return *reinterpret_cast<unsigned*>(&h);
}

// ---------------------------------------------------------------------------
// fill (grid-stride)
// ---------------------------------------------------------------------------
__global__ void fill_kernel(float* __restrict__ p, float v, size_t n) {
  size_t i = (size_t)blockIdx.x * blockDim.x + threadIdx.x;
  size_t st = (size_t)gridDim.x * blockDim.x;
  for (; i < n; i += st) p[i] = v;
}

// ---------------------------------------------------------------------------
// GEMM: Y[M,N] = act(X[M,K] @ W[K,N] + bias[N]) via v_wmma_f32_16x16x32_bf16
// REQUIRES K % 32 == 0 (true for every call site in this network).
// Block tile 64x64, 128 threads = 4 waves, each wave owns a 32x32 quadrant
// (2x2 WMMA tiles). A staged row-major, W staged K-transposed so both
// fragment reads are contiguous ds_load_b128 per ISA 7.12.2 layouts.
// Staging: clause of global_load_b128, packed v_cvt_pk_bf16_f32, vector LDS
// stores (A: 2x b128; B: 8x b32 carrying a K-pair each).
// ---------------------------------------------------------------------------
__global__ __launch_bounds__(128)
void gemm_kernel(const float* __restrict__ X, const float* __restrict__ W,
                 const float* __restrict__ bias, float* __restrict__ Y,
                 int M, int N, int K, int act)
{
  __shared__ __bf16 As[64][40];   // [row][k 0..31] (+pad); row stride 80B
  __shared__ __bf16 Bt[64][40];   // [col][k 0..31] (+pad)

  const int tid  = threadIdx.x;
  const int lane = tid & 31;
  const int wave = tid >> 5;          // 0..3
  const int m0b  = blockIdx.x * 64;
  const int n0b  = blockIdx.y * 64;
  const int wm   = (wave >> 1) * 32;
  const int wn   = (wave & 1) * 32;
  const int lr   = lane & 15;
  const int lh   = lane >> 4;

  v8f acc[2][2] = {};

  // A staging map: row = tid>>1 (0..63), 16-wide K half = tid&1
  const int arow  = tid >> 1;
  const int ahalf = tid & 1;
  const int gm    = m0b + arow;
  // B staging map: K pair = tid>>3 (0..15), 8-wide column group = (tid&7)*8
  const int bkp = (tid >> 3) * 2;     // k offset within tile: 0,2,..,30
  const int bcg = (tid & 7) * 8;
  const int gn0 = n0b + bcg;
  const bool bfull = (gn0 + 8 <= N);

  for (int k0 = 0; k0 < K; k0 += 32) {
    // ---- stage A tile: 4x float4 loads -> 8 packed dwords -> 2x b128 ----
    {
      float4 f0 = {0,0,0,0}, f1 = {0,0,0,0}, f2 = {0,0,0,0}, f3 = {0,0,0,0};
      if (gm < M) {
        const float4* px =
            reinterpret_cast<const float4*>(&X[(size_t)gm * K + k0 + ahalf * 16]);
        f0 = px[0]; f1 = px[1]; f2 = px[2]; f3 = px[3];
        if (k0 + 32 < K)
          __builtin_prefetch(&X[(size_t)gm * K + k0 + 32 + ahalf * 16], 0, 3);
      }
      uint4* da = reinterpret_cast<uint4*>(&As[arow][ahalf * 16]);
      da[0] = make_uint4(pkbf(f0.x, f0.y), pkbf(f0.z, f0.w),
                         pkbf(f1.x, f1.y), pkbf(f1.z, f1.w));
      da[1] = make_uint4(pkbf(f2.x, f2.y), pkbf(f2.z, f2.w),
                         pkbf(f3.x, f3.y), pkbf(f3.z, f3.w));
    }
    // ---- stage W tile transposed: rows kk,kk+1 x 8 cols -> 8x b32 ----
    {
      const int kk = k0 + bkp;         // kk+1 < K guaranteed (K % 32 == 0)
      float r0[8], r1[8];
      if (bfull) {
        const float4* p0 = reinterpret_cast<const float4*>(&W[(size_t)kk * N + gn0]);
        const float4* p1 =
            reinterpret_cast<const float4*>(&W[(size_t)(kk + 1) * N + gn0]);
        float4 a0 = p0[0], a1 = p0[1], c0 = p1[0], c1 = p1[1];
        r0[0]=a0.x; r0[1]=a0.y; r0[2]=a0.z; r0[3]=a0.w;
        r0[4]=a1.x; r0[5]=a1.y; r0[6]=a1.z; r0[7]=a1.w;
        r1[0]=c0.x; r1[1]=c0.y; r1[2]=c0.z; r1[3]=c0.w;
        r1[4]=c1.x; r1[5]=c1.y; r1[6]=c1.z; r1[7]=c1.w;
      } else {
#pragma unroll
        for (int i = 0; i < 8; ++i) {
          const bool ok = (gn0 + i < N);
          r0[i] = ok ? W[(size_t)kk * N + gn0 + i] : 0.0f;
          r1[i] = ok ? W[(size_t)(kk + 1) * N + gn0 + i] : 0.0f;
        }
      }
#pragma unroll
      for (int i = 0; i < 8; ++i)
        *reinterpret_cast<unsigned*>(&Bt[bcg + i][bkp]) = pkbf(r0[i], r1[i]);
    }
    __syncthreads();

    // ---- fragments (contiguous 16B LDS reads per ISA 7.12.2) ----
    v16bf afrag[2], bfrag[2];
#pragma unroll
    for (int t = 0; t < 2; ++t) {
      const int row = wm + t * 16 + lr;
      const __bf16* pa = &As[row][lh * 8];
#pragma unroll
      for (int i = 0; i < 8; ++i) afrag[t][i] = pa[i];
      pa = &As[row][16 + lh * 8];
#pragma unroll
      for (int i = 0; i < 8; ++i) afrag[t][8 + i] = pa[i];

      const int col = wn + t * 16 + lr;
      const __bf16* pb = &Bt[col][lh * 8];
#pragma unroll
      for (int i = 0; i < 8; ++i) bfrag[t][i] = pb[i];
      pb = &Bt[col][16 + lh * 8];
#pragma unroll
      for (int i = 0; i < 8; ++i) bfrag[t][8 + i] = pb[i];
    }

#pragma unroll
    for (int ti = 0; ti < 2; ++ti)
#pragma unroll
      for (int tj = 0; tj < 2; ++tj)
        acc[ti][tj] = __builtin_amdgcn_wmma_f32_16x16x32_bf16(
            false, afrag[ti], false, bfrag[tj],
            (short)0, acc[ti][tj], false, false);
    __syncthreads();
  }

  // ---- epilogue: bias + optional relu, guarded stores (C/D layout) ----
#pragma unroll
  for (int ti = 0; ti < 2; ++ti)
#pragma unroll
    for (int tj = 0; tj < 2; ++tj) {
      const int col = n0b + wn + tj * 16 + lr;
      if (col < N) {
        const float bv = bias[col];
#pragma unroll
        for (int r = 0; r < 8; ++r) {
          const int row = m0b + wm + ti * 16 + lh * 8 + r;
          if (row < M) {
            float v = acc[ti][tj][r] + bv;
            if (act) v = fmaxf(v, 0.0f);
            Y[(size_t)row * N + col] = v;
          }
        }
      }
    }
}

// ---------------------------------------------------------------------------
// Edge pass 1: msg = relu(x'[src] + edge_attr@We + be) + EPS ; segment max
// (atomicMax on float bits: all msgs > 0, bit pattern is order-preserving)
// ---------------------------------------------------------------------------
__global__ __launch_bounds__(256)
void edge_msg_max_kernel(const float* __restrict__ xp, const float* __restrict__ ea,
                         const float* __restrict__ We, const float* __restrict__ be,
                         const int* __restrict__ src, const int* __restrict__ dst,
                         float* __restrict__ mmax, int E, int C)
{
  __shared__ float eash[4][16];
  const int ty = threadIdx.x >> 6;
  const int tx = threadIdx.x & 63;
  const int e  = blockIdx.x * 4 + ty;
  if (tx < 16 && e < E) eash[ty][tx] = ea[(size_t)e * 16 + tx];
  __syncthreads();
  if (e >= E) return;
  const int s = src[e], d = dst[e];
  for (int f = tx; f < C; f += 64) {
    float ev = be[f];
#pragma unroll
    for (int k = 0; k < 16; ++k) ev = fmaf(eash[ty][k], We[k * C + f], ev);
    const float msg = fmaxf(xp[(size_t)s * C + f] + ev, 0.0f) + 1e-7f;
    atomicMax((int*)&mmax[(size_t)d * C + f], __float_as_int(msg));
  }
}

// ---------------------------------------------------------------------------
// Edge pass 2: ex = exp(msg - max[dst]); s += ex; w += ex*msg
// ---------------------------------------------------------------------------
__global__ __launch_bounds__(256)
void edge_softmax_kernel(const float* __restrict__ xp, const float* __restrict__ ea,
                         const float* __restrict__ We, const float* __restrict__ be,
                         const int* __restrict__ src, const int* __restrict__ dst,
                         const float* __restrict__ mmax,
                         float* __restrict__ ssum, float* __restrict__ wsum,
                         int E, int C)
{
  __shared__ float eash[4][16];
  const int ty = threadIdx.x >> 6;
  const int tx = threadIdx.x & 63;
  const int e  = blockIdx.x * 4 + ty;
  if (tx < 16 && e < E) eash[ty][tx] = ea[(size_t)e * 16 + tx];
  __syncthreads();
  if (e >= E) return;
  const int s = src[e], d = dst[e];
  for (int f = tx; f < C; f += 64) {
    float ev = be[f];
#pragma unroll
    for (int k = 0; k < 16; ++k) ev = fmaf(eash[ty][k], We[k * C + f], ev);
    const float msg = fmaxf(xp[(size_t)s * C + f] + ev, 0.0f) + 1e-7f;
    const float ex  = __expf(msg - mmax[(size_t)d * C + f]);
    atomicAdd(&ssum[(size_t)d * C + f], ex);
    atomicAdd(&wsum[(size_t)d * C + f], ex * msg);
  }
}

// h2 = x' + wsum / (ssum + 1e-16)
__global__ void node_update_kernel(const float* __restrict__ xp,
                                   const float* __restrict__ wsum,
                                   const float* __restrict__ ssum,
                                   float* __restrict__ out, size_t n)
{
  size_t i = (size_t)blockIdx.x * blockDim.x + threadIdx.x;
  if (i < n) out[i] = xp[i] + wsum[i] / (ssum[i] + 1e-16f);
}

// ---------------------------------------------------------------------------
// BatchNorm (training mode, biased var): per-feature column reduction
// ---------------------------------------------------------------------------
__global__ __launch_bounds__(256)
void bn_stats_kernel(const float* __restrict__ h, float* __restrict__ mean,
                     float* __restrict__ var, int M, int C)
{
  __shared__ float s1[256], s2[256];
  const int f = blockIdx.x;
  float a = 0.0f, b = 0.0f;
  for (int r = threadIdx.x; r < M; r += 256) {
    const float v = h[(size_t)r * C + f];
    a += v; b += v * v;
  }
  s1[threadIdx.x] = a; s2[threadIdx.x] = b;
  __syncthreads();
  for (int st = 128; st > 0; st >>= 1) {
    if (threadIdx.x < st) {
      s1[threadIdx.x] += s1[threadIdx.x + st];
      s2[threadIdx.x] += s2[threadIdx.x + st];
    }
    __syncthreads();
  }
  if (threadIdx.x == 0) {
    const float mu = s1[0] / (float)M;
    mean[f] = mu;
    var[f]  = s2[0] / (float)M - mu * mu;
  }
}

__global__ void bn_apply_relu_kernel(float* __restrict__ h,
                                     const float* __restrict__ mean,
                                     const float* __restrict__ var,
                                     const float* __restrict__ gamma,
                                     const float* __restrict__ beta,
                                     int M, int C)
{
  size_t i = (size_t)blockIdx.x * blockDim.x + threadIdx.x;
  if (i >= (size_t)M * C) return;
  const int f = (int)(i % C);
  const float v = (h[i] - mean[f]) * rsqrtf(var[f] + 1e-5f) * gamma[f] + beta[f];
  h[i] = fmaxf(v, 0.0f);
}

// ---------------------------------------------------------------------------
// Global mean pool
// ---------------------------------------------------------------------------
__global__ void pool_accum_kernel(const float* __restrict__ h,
                                  const int* __restrict__ batch,
                                  float* __restrict__ pool, int M, int C)
{
  size_t i = (size_t)blockIdx.x * blockDim.x + threadIdx.x;
  if (i >= (size_t)M * C) return;
  const int n = (int)(i / C), f = (int)(i % C);
  const int b = batch[n];
  if (b >= 0 && b < NGRAPHS) atomicAdd(&pool[(size_t)b * C + f], h[i]);
}

__global__ void count_kernel(const int* __restrict__ batch, float* __restrict__ cnt, int M)
{
  int i = blockIdx.x * blockDim.x + threadIdx.x;
  if (i < M) {
    const int b = batch[i];
    if (b >= 0 && b < NGRAPHS) atomicAdd(&cnt[b], 1.0f);
  }
}

__global__ void pool_div_kernel(float* __restrict__ pool, const float* __restrict__ cnt,
                                int G, int C)
{
  int i = blockIdx.x * blockDim.x + threadIdx.x;
  if (i < G * C) pool[i] /= fmaxf(cnt[i / C], 1.0f);
}

// ---------------------------------------------------------------------------
// log_softmax over rows
// ---------------------------------------------------------------------------
__global__ void log_softmax_kernel(const float* __restrict__ gin,
                                   float* __restrict__ out, int G, int N)
{
  int r = blockIdx.x * blockDim.x + threadIdx.x;
  if (r >= G) return;
  float mx = -1e30f;
  for (int j = 0; j < N; ++j) mx = fmaxf(mx, gin[r * N + j]);
  float s = 0.0f;
  for (int j = 0; j < N; ++j) s += __expf(gin[r * N + j] - mx);
  const float lse = mx + logf(s);
  for (int j = 0; j < N; ++j) out[r * N + j] = gin[r * N + j] - lse;
}

// ---------------------------------------------------------------------------
// Host orchestration
// ---------------------------------------------------------------------------
extern "C" void kernel_launch(void* const* d_in, const int* in_sizes, int n_in,
                              void* d_out, int out_size, void* d_ws, size_t ws_size,
                              hipStream_t stream)
{
  (void)in_sizes; (void)n_in; (void)out_size; (void)ws_size;

  const float* x  = (const float*)d_in[0];
  const float* ea = (const float*)d_in[1];

  // params flattened alphabetically per dict (jax tree order):
  // conv with Ws : W1 W2 We Ws b1 b2 be be1 bs g1   (10 leaves)
  // conv w/o Ws  : W1 W2 We b1 b2 be be1 g1          (8 leaves)
  // dense        : W b
  struct ConvP { const float *W1,*W2,*We,*Ws,*b1,*b2,*be,*be1,*bs,*g1; int cin,cout; };
  const int dims[6] = {128, 64, 64, 128, 256, 512};
  ConvP cp[5];
  int idx = 2;
  for (int L = 0; L < 5; ++L) {
    const bool hasWs = (dims[L] != dims[L + 1]);
    cp[L].cin = dims[L]; cp[L].cout = dims[L + 1];
    cp[L].W1 = (const float*)d_in[idx++];
    cp[L].W2 = (const float*)d_in[idx++];
    cp[L].We = (const float*)d_in[idx++];
    cp[L].Ws = hasWs ? (const float*)d_in[idx++] : nullptr;
    cp[L].b1 = (const float*)d_in[idx++];
    cp[L].b2 = (const float*)d_in[idx++];
    cp[L].be = (const float*)d_in[idx++];
    cp[L].be1 = (const float*)d_in[idx++];
    cp[L].bs = hasWs ? (const float*)d_in[idx++] : nullptr;
    cp[L].g1 = (const float*)d_in[idx++];
  }
  const float* d1W = (const float*)d_in[idx++]; const float* d1b = (const float*)d_in[idx++];
  const float* d2W = (const float*)d_in[idx++]; const float* d2b = (const float*)d_in[idx++];
  const float* d3W = (const float*)d_in[idx++]; const float* d3b = (const float*)d_in[idx++];
  const int* eidx  = (const int*)d_in[idx++];   // [2, E]
  const int* batch = (const int*)d_in[idx++];   // [N]
  const int* src = eidx;
  const int* dst = eidx + NEDGES;

  // workspace carve-up (padded to 16-float = 64B boundaries for float4 loads)
  float* wsf = (float*)d_ws;
  size_t off = 0;
  auto alloc = [&](size_t nf) {
    float* p = wsf + off;
    off += (nf + 15) & ~(size_t)15;
    return p;
  };
  float* hA   = alloc((size_t)NNODES * 512);   // layer output / next input
  float* hB   = alloc((size_t)NNODES * 512);   // x' = lin_src(x)
  float* hC   = alloc((size_t)NNODES * 1024);  // MLP hidden (2*cout)
  float* hD   = alloc((size_t)NNODES * 512);   // x' + agg
  float* mbuf = alloc((size_t)NNODES * 512);   // segment max
  float* sbuf = alloc((size_t)NNODES * 512);   // softmax denom
  float* wbuf = alloc((size_t)NNODES * 512);   // softmax numer
  float* mean = alloc(1024);
  float* var  = alloc(1024);
  float* pool = alloc((size_t)NGRAPHS * 512);
  float* cnt  = alloc(NGRAPHS);
  float* g1b  = alloc((size_t)NGRAPHS * 512);
  float* g2b  = alloc((size_t)NGRAPHS * 256);
  float* g3b  = alloc((size_t)NGRAPHS * 16);

  auto gemm = [&](const float* Xp, const float* Wp, const float* bp, float* Yp,
                  int M, int N, int K, int act) {
    dim3 g((M + 63) / 64, (N + 63) / 64);
    gemm_kernel<<<g, 128, 0, stream>>>(Xp, Wp, bp, Yp, M, N, K, act);
  };
  auto fill = [&](float* p, float v, size_t n) {
    fill_kernel<<<2048, 256, 0, stream>>>(p, v, n);
  };

  const int M = NNODES, E = NEDGES;
  for (int L = 0; L < 5; ++L) {
    const int cin = cp[L].cin, cout = cp[L].cout;
    const float* xin = (L == 0) ? x : hA;
    const float* xp;
    if (cp[L].Ws) { gemm(xin, cp[L].Ws, cp[L].bs, hB, M, cout, cin, 0); xp = hB; }
    else xp = xin;

    const size_t nc = (size_t)M * cout;
    fill(mbuf, 0.0f, nc);
    fill(sbuf, 0.0f, nc);
    fill(wbuf, 0.0f, nc);

    const int eblocks = (E + 3) / 4;
    edge_msg_max_kernel<<<eblocks, 256, 0, stream>>>(xp, ea, cp[L].We, cp[L].be,
                                                     src, dst, mbuf, E, cout);
    edge_softmax_kernel<<<eblocks, 256, 0, stream>>>(xp, ea, cp[L].We, cp[L].be,
                                                     src, dst, mbuf, sbuf, wbuf, E, cout);
    node_update_kernel<<<(int)((nc + 255) / 256), 256, 0, stream>>>(xp, wbuf, sbuf, hD, nc);

    const int c2 = 2 * cout;
    gemm(hD, cp[L].W1, cp[L].b1, hC, M, c2, cout, 0);
    bn_stats_kernel<<<c2, 256, 0, stream>>>(hC, mean, var, M, c2);
    const size_t n2 = (size_t)M * c2;
    bn_apply_relu_kernel<<<(int)((n2 + 255) / 256), 256, 0, stream>>>(hC, mean, var,
                                                                      cp[L].g1, cp[L].be1, M, c2);
    gemm(hC, cp[L].W2, cp[L].b2, hA, M, cout, c2, 1);  // fused layer relu
  }

  // global mean pool (hA is [N, 512])
  fill(pool, 0.0f, (size_t)NGRAPHS * 512);
  fill(cnt, 0.0f, NGRAPHS);
  const size_t np = (size_t)M * 512;
  pool_accum_kernel<<<(int)((np + 255) / 256), 256, 0, stream>>>(hA, batch, pool, M, 512);
  count_kernel<<<(M + 255) / 256, 256, 0, stream>>>(batch, cnt, M);
  pool_div_kernel<<<(NGRAPHS * 512 + 255) / 256, 256, 0, stream>>>(pool, cnt, NGRAPHS, 512);

  // dense head
  gemm(pool, d1W, d1b, g1b, NGRAPHS, 512, 512, 0);
  gemm(g1b,  d2W, d2b, g2b, NGRAPHS, 256, 512, 0);
  gemm(g2b,  d3W, d3b, g3b, NGRAPHS, 10, 256, 0);
  log_softmax_kernel<<<1, 64, 0, stream>>>(g3b, (float*)d_out, NGRAPHS, 10);
}
